// ProductQuantizer_38465727103638
// MI455X (gfx1250) — matrix operand order repaired and checked by reading, not compiled, not run on hardware
//
#include <hip/hip_runtime.h>
#include <hip/hip_bf16.h>

// Problem constants (match reference)
#define B_   16
#define T_   512
#define D_   1024
#define S_   4
#define K_   1024
#define DS_  256
#define N_ROWS (B_ * T_)            // 8192 vectors
#define ROWS_PER_BLOCK 128
#define NTILES  (K_ / 16)           // 64 centroid tiles
#define KCHUNKS (DS_ / 32)          // 8 WMMA k-steps per tile
#define TILE_ELEMS (KCHUNKS * 32 * 16)   // 4096 bf16 = 8 KB per B tile

typedef __attribute__((ext_vector_type(16))) __bf16 v16bf;
typedef __attribute__((ext_vector_type(8)))  float  v8f;

union FragU {
    uint4 q[2];
    v16bf v;
    unsigned short u[16];
};

__device__ __forceinline__ unsigned short f2bf(float f) {
    // round-to-nearest-even fp32 -> bf16
    unsigned int u = __float_as_uint(f);
    return (unsigned short)((u + 0x7FFFu + ((u >> 16) & 1u)) >> 16);
}

// Generic pointers to LDS carry the wave-relative LDS byte offset in their
// low 32 bits (ISA: flat->LDS uses addr[31:0]); async-to-LDS VDST wants that.
__device__ __forceinline__ unsigned lds_off(const void* p) {
    return (unsigned)(unsigned long long)p;
}

// Asynchronously copy one 8 KB B-fragment tile (contiguous in global) into
// LDS, spread across all 256 threads (32 B each). Tracked by ASYNCcnt.
// INST_OFFSET is added to both the global and the LDS address, so one
// offset:16 covers the second 16-byte half.
__device__ __forceinline__ void stage_async_tile(const unsigned short* __restrict__ gsrc,
                                                 unsigned short* lds_dst, int t) {
    unsigned loff = lds_off(lds_dst) + (unsigned)t * 32u;
    const unsigned short* g = gsrc + (size_t)t * 16;
    asm volatile(
        "global_load_async_to_lds_b128 %0, %1, off\n\t"
        "global_load_async_to_lds_b128 %0, %1, off offset:16"
        :: "v"(loff), "v"(g)
        : "memory");
}

__device__ __forceinline__ void wait_async_zero() {
    asm volatile("s_wait_asynccnt 0x0" ::: "memory");
}

// ---------------------------------------------------------------------------
// Kernel 1: repack codebooks (fp32 [S][K][DS]) into bf16 WMMA B-fragments.
// Layout: frag[s][ntile][kc][lane][e], lane holds column n = ntile*16 + lane%16,
// elements e=0..15 map to K = kc*32 + (lane>=16 ? 16 : 0) + e  (contiguous; same order as A).
// One wave per (s, ntile, kc): 4*64*8 = 2048 waves.
// ---------------------------------------------------------------------------
__global__ __launch_bounds__(256) void pq_pack_cb(const float* __restrict__ cb,
                                                  unsigned short* __restrict__ frag) {
    int w    = (blockIdx.x * 256 + threadIdx.x) >> 5;   // global wave id 0..2047
    int lane = threadIdx.x & 31;
    int s     = w >> 9;         // / (64*8)
    int rem   = w & 511;
    int ntile = rem >> 3;
    int kc    = rem & 7;

    int n     = ntile * 16 + (lane & 15);
    int kbase = kc * 32 + ((lane >> 4) << 4);           // +0 or +16

    const float* src = cb + ((size_t)(s * K_ + n) * DS_ + kbase);
    FragU f;
#pragma unroll
    for (int e = 0; e < 16; ++e) f.u[e] = f2bf(src[e]);

    uint4* dst = (uint4*)(frag + ((size_t)w * 32 + lane) * 16);
    dst[0] = f.q[0];
    dst[1] = f.q[1];
}

// ---------------------------------------------------------------------------
// Kernel 2: c2[s][k] = ||codebook row||^2 , and passthrough-copy codebooks
// to output slot 4. One 256-thread block per (s,k) row.
// ---------------------------------------------------------------------------
__global__ __launch_bounds__(256) void pq_c2_copy(const float* __restrict__ cb,
                                                  float* __restrict__ c2,
                                                  float* __restrict__ out_cb) {
    int row = blockIdx.x;               // 0..4095
    int t   = threadIdx.x;              // 0..255
    float v = cb[(size_t)row * DS_ + t];
    out_cb[(size_t)row * DS_ + t] = v;

    __shared__ float red[256];
    red[t] = v * v;
    __syncthreads();
#pragma unroll
    for (int o = 128; o > 0; o >>= 1) {
        if (t < o) red[t] += red[t + o];
        __syncthreads();
    }
    if (t == 0) c2[row] = red[0];
}

// ---------------------------------------------------------------------------
// Kernel 3: main PQ search + gather.
// Grid: 4 sections * 64 row-tiles = 256 blocks of 256 threads (8 waves).
// Each wave owns 16 rows; A-fragments stay in registers for the whole N loop.
// B tiles are async-copied into LDS once per workgroup (double-buffered) and
// shared by all 8 waves. Score = c2[k] - 2*dot (x^2 drops out of argmin).
// ---------------------------------------------------------------------------
__global__ __launch_bounds__(256) void pq_main_kernel(
        const float* __restrict__ x, const float* __restrict__ cb,
        const unsigned short* __restrict__ frag, const float* __restrict__ c2,
        float* __restrict__ out_q, float* __restrict__ out_loss,
        float* __restrict__ out_nn) {
    __shared__ unsigned short xs[ROWS_PER_BLOCK * DS_];   // 64 KB bf16 x-tile
    __shared__ float c2s[K_];                             // 4 KB
    __shared__ unsigned short bs[2][TILE_ELEMS];          // 2 x 8 KB B tiles

    const int s       = blockIdx.x >> 6;   // section 0..3
    const int tile    = blockIdx.x & 63;
    const int rowBase = tile * ROWS_PER_BLOCK;
    const int t       = threadIdx.x;

    const unsigned short* fsec = frag + (size_t)s * NTILES * TILE_ELEMS;

    // kick off async copy of B tile 0 immediately (overlaps with x staging)
    stage_async_tile(fsec, &bs[0][0], t);

    // stage ||c||^2 for this section
    for (int i = t; i < K_; i += 256) c2s[i] = c2[s * K_ + i];

    // stage x-tile: fp32 -> bf16 into LDS, coalesced float4 loads
#pragma unroll
    for (int it = 0; it < 32; ++it) {
        int idx = it * 256 + t;            // 0..8191
        int row = idx >> 6;                // 0..127
        int c4  = idx & 63;                // float4 column
        float4 v = *(const float4*)(x + ((size_t)(rowBase + row) * D_ + s * DS_ + c4 * 4));
        uint2 pk;
        pk.x = (unsigned)f2bf(v.x) | ((unsigned)f2bf(v.y) << 16);
        pk.y = (unsigned)f2bf(v.z) | ((unsigned)f2bf(v.w) << 16);
        *(uint2*)&xs[row * DS_ + c4 * 4] = pk;
    }
    __syncthreads();

    const int wave = t >> 5;
    const int lane = t & 31;
    const int half = lane >> 4;            // 0 | 1
    const int nl   = lane & 15;
    const int mRow = wave * 16 + nl;       // local A row for this lane

    // Build A fragments (resident for whole N loop).
    // A k order used here (same as B pack): lane<16 -> K {0..15}; lane>=16 -> {16..31}
    v16bf a[KCHUNKS];
    const int koff = half * 16;
#pragma unroll
    for (int kc = 0; kc < KCHUNKS; ++kc) {
        FragU f;
        int k0 = kc * 32 + koff;
        f.q[0] = *(const uint4*)&xs[mRow * DS_ + k0];        // K = k0..k0+7
        f.q[1] = *(const uint4*)&xs[mRow * DS_ + k0 + 8];    // K = k0+8..k0+15
        a[kc] = f.v;
    }

    float best[8];
    int   bidx[8];
#pragma unroll
    for (int r = 0; r < 8; ++r) { best[r] = 3.4e38f; bidx[r] = 0; }

    for (int nt = 0; nt < NTILES; ++nt) {
        // wait for this wave's async share of the current tile, then make the
        // whole tile visible across waves
        wait_async_zero();
        __syncthreads();

        // prefetch next tile into the other buffer (async, overlaps compute);
        // safe: every wave finished reading that buffer before its previous
        // barrier (its ds reads were consumed by the WMMAs of iteration nt-1)
        if (nt + 1 < NTILES)
            stage_async_tile(fsec + (size_t)(nt + 1) * TILE_ELEMS,
                             &bs[(nt + 1) & 1][0], t);

        const unsigned short* fb = &bs[nt & 1][lane * 16];
        v16bf b[KCHUNKS];
#pragma unroll
        for (int kc = 0; kc < KCHUNKS; ++kc) {
            FragU f;
            f.q[0] = *(const uint4*)(fb + kc * 32 * 16);
            f.q[1] = *(const uint4*)(fb + kc * 32 * 16 + 8);
            b[kc] = f.v;
        }

        v8f c = {0.f, 0.f, 0.f, 0.f, 0.f, 0.f, 0.f, 0.f};
#pragma unroll
        for (int kc = 0; kc < KCHUNKS; ++kc)
            c = __builtin_amdgcn_wmma_f32_16x16x32_bf16(
                    false, a[kc], false, b[kc], (short)0, c, false, false);

        // C layout: element r -> row m = half*8 + r, col n = nt*16 + nl
        int n = nt * 16 + nl;
        float c2n = c2s[n];
#pragma unroll
        for (int r = 0; r < 8; ++r) {
            float score = fmaf(-2.0f, c[r], c2n);
            if (score < best[r]) { best[r] = score; bidx[r] = n; }
        }
    }

    // Cross-lane argmin: row m lives across the 16 lanes of its half-wave.
#pragma unroll
    for (int off = 1; off < 16; off <<= 1) {
#pragma unroll
        for (int r = 0; r < 8; ++r) {
            float ob = __shfl_xor(best[r], off, 32);
            int   oi = __shfl_xor(bidx[r], off, 32);
            if (ob < best[r] || (ob == best[r] && oi < bidx[r])) {
                best[r] = ob;
                bidx[r] = oi;
            }
        }
    }

    // Broadcast winners to all 32 lanes (lane 0 holds m=0..7, lane 16 m=8..15)
    int wiLo[8], wiHi[8];
#pragma unroll
    for (int r = 0; r < 8; ++r) {
        wiLo[r] = __shfl(bidx[r], 0, 32);
        wiHi[r] = __shfl(bidx[r], 16, 32);
    }

    // Gather exact fp32 codewords + write quantized / loss, coalesced.
    const int gmBase = rowBase + wave * 16;
#pragma unroll
    for (int m = 0; m < 16; ++m) {
        int kidx = (m < 8) ? wiLo[m] : wiHi[m - 8];
        int gm = gmBase + m;
        const float4* qrow = (const float4*)(cb + ((size_t)(s * K_ + kidx) * DS_));
        const float4* xrow = (const float4*)(x + ((size_t)gm * D_ + s * DS_));
        float4* oq = (float4*)(out_q    + ((size_t)gm * D_ + s * DS_));
        float4* ol = (float4*)(out_loss + ((size_t)gm * D_ + s * DS_));
#pragma unroll
        for (int j = 0; j < 2; ++j) {
            int c4 = lane * 2 + j;              // 0..63
            float4 q4 = qrow[c4];
            float4 x4 = xrow[c4];
            oq[c4] = q4;
            float4 l4;
            l4.x = (q4.x - x4.x) * (q4.x - x4.x);
            l4.y = (q4.y - x4.y) * (q4.y - x4.y);
            l4.z = (q4.z - x4.z) * (q4.z - x4.z);
            l4.w = (q4.w - x4.w) * (q4.w - x4.w);
            ol[c4] = l4;
        }
    }

    // Indices: lane 0 writes m=0..7, lane 16 writes m=8..15 (static indexing)
    if (nl == 0) {
#pragma unroll
        for (int r = 0; r < 8; ++r)
            out_nn[s * N_ROWS + gmBase + half * 8 + r] = (float)bidx[r];
    }
}

// ---------------------------------------------------------------------------
extern "C" void kernel_launch(void* const* d_in, const int* in_sizes, int n_in,
                              void* d_out, int out_size, void* d_ws, size_t ws_size,
                              hipStream_t stream) {
    const float* x  = (const float*)d_in[0];   // [B,T,D] fp32
    const float* cb = (const float*)d_in[1];   // [S,K,DS] fp32

    float* out = (float*)d_out;
    const size_t QN = (size_t)B_ * T_ * D_;    // 8,388,608
    float* out_q    = out;                              // quantized
    float* out_loss = out + QN;                         // (q - x)^2
    float* out_nn   = out + 2 * QN;                     // indices [S, B*T]
    float* out_cb   = out + 2 * QN + (size_t)S_ * N_ROWS; // codebooks passthrough

    unsigned short* frag = (unsigned short*)d_ws;                    // 2 MB bf16 fragments
    float* c2 = (float*)((char*)d_ws + (size_t)2 * 1024 * 1024);     // 16 KB

    // 1) repack codebooks into bf16 WMMA B-fragments (2048 waves)
    pq_pack_cb<<<256, 256, 0, stream>>>(cb, frag);
    // 2) centroid norms + codebook passthrough
    pq_c2_copy<<<S_ * K_, 256, 0, stream>>>(cb, c2, out_cb);
    // 3) fused distance-GEMM (WMMA) + argmin + gather, async B staging
    pq_main_kernel<<<S_ * (N_ROWS / ROWS_PER_BLOCK), 256, 0, stream>>>(
        x, cb, frag, c2, out_q, out_loss, out_nn);
}
